// ScannedRNN_40011915329857
// MI455X (gfx1250) — matrix operand order repaired and checked by reading, not compiled, not run on hardware
//
#include <hip/hip_runtime.h>
#include <hip/hip_bf16.h>
#include <stdint.h>
#include <stddef.h>

// ---------------------------------------------------------------------------
// GRU scan, T=512, B=64, H=512.  bf16 WMMA (f32 accum) on gfx1250.
// ---------------------------------------------------------------------------

typedef __bf16 bf16;
typedef bf16  v16bf __attribute__((ext_vector_type(16)));
typedef bf16  v8bf  __attribute__((ext_vector_type(8)));
typedef float v8f   __attribute__((ext_vector_type(8)));

#define T_STEPS 512
#define BATCH   64
#define HID     512
#define NWG     8            // persistent workgroups in scan (one per WGP)
#define JW      64           // hidden units per workgroup slice
#define WPITCH  520          // LDS weight row pitch (elems); 260 dwords -> bank-conflict free
#define SPITCH  196          // LDS fp32 scratch pitch

static __device__ __forceinline__ v8f wmma_bf16(v16bf a, v16bf b, v8f c) {
  return __builtin_amdgcn_wmma_f32_16x16x32_bf16(false, a, false, b, (short)0, c,
                                                 false, false);
}

static __device__ __forceinline__ v16bf frag_join(const bf16* p0, const bf16* p1) {
  union { v16bf v; v8bf h[2]; } u;
  u.h[0] = *(const v8bf*)p0;
  u.h[1] = *(const v8bf*)p1;
  return u.v;
}

// A fragment 16x32 bf16 (row-major source, row pitch `pitch` elems).
// lanes 0-15: row M=l, K = {0..7, 16..23}; lanes 16-31: row M=l-16, K = {8..15, 24..31}
static __device__ __forceinline__ v16bf load_a_frag(const bf16* base, int row0,
                                                    int kb, int lane, int pitch) {
  int l16 = lane & 15;
  int hi  = (lane >> 4) & 1;
  const bf16* p = base + (size_t)(row0 + l16) * pitch + kb * 32 + hi * 8;
  return frag_join(p, p + 16);
}

// B fragment 32x16 bf16 from transposed weights wt[n][k] (k contiguous).
// lanes 0-15: col N=l, K = {0..7, 16..23}; lanes 16-31: col N=l-16, K = {8..15, 24..31}
static __device__ __forceinline__ v16bf load_b_frag(const bf16* wt, int col0,
                                                    int kb, int lane, int pitch) {
  int l16 = lane & 15;
  int hi  = (lane >> 4) & 1;
  const bf16* p = wt + (size_t)(col0 + l16) * pitch + kb * 32 + hi * 8;
  return frag_join(p, p + 16);
}

// ---------------------------------------------------------------------------
// K0: zero h-state (fp32 + bf16 mirror) and barrier counter
__global__ void k_init(float* hstate, bf16* hmask, unsigned int* cnt) {
  int i = blockIdx.x * blockDim.x + threadIdx.x;
  if (i == 0) *cnt = 0u;
  if (i < BATCH * HID) {
    hstate[i] = 0.f;
    hmask[i]  = (bf16)0.f;
  }
}

// K1: ins fp32 -> bf16
__global__ void k_cvt_ins(const float* __restrict__ ins, bf16* __restrict__ insb, int n) {
  int i = blockIdx.x * blockDim.x + threadIdx.x;
  if (i < n) insb[i] = (bf16)ins[i];
}

// K2: W_i [512,1536] -> bf16 transposed wt[n][k]
__global__ void k_build_wti(const float* __restrict__ W_i, bf16* __restrict__ wt) {
  int idx = blockIdx.x * blockDim.x + threadIdx.x;
  if (idx < 3 * HID * HID) {
    int n = idx / HID, k = idx % HID;
    wt[idx] = (bf16)W_i[(size_t)k * (3 * HID) + n];
  }
}

// K3: recurrent weights -> bf16 transposed wr[row][k]
//   rows 0..511   : W_hrz col j        (r gate)
//   rows 512..1023: W_hrz col 512+j    (z gate)
//   rows 1024..   : W_hn  col j        (n gate)
__global__ void k_build_wrec(const float* __restrict__ W_hrz,
                             const float* __restrict__ W_hn,
                             bf16* __restrict__ wr) {
  int idx = blockIdx.x * blockDim.x + threadIdx.x;
  if (idx < 3 * HID * HID) {
    int row = idx / HID, k = idx % HID;
    float v = (row < 2 * HID) ? W_hrz[(size_t)k * (2 * HID) + row]
                              : W_hn[(size_t)k * HID + (row - 2 * HID)];
    wr[idx] = (bf16)v;
  }
}

// ---------------------------------------------------------------------------
// K4: xp = insb @ W_i + b_i   ([32768,512] x [512,1536], fp32 out)
// one wave computes a 16x64 strip; 2048 row blocks x 24 col groups = 49152 waves
__global__ void k_gemm_xp(const bf16* __restrict__ insb, const bf16* __restrict__ wt,
                          const float* __restrict__ b_i, float* __restrict__ xp) {
  int gwave = (blockIdx.x * blockDim.x + threadIdx.x) >> 5;
  int lane  = threadIdx.x & 31;
  int mblk  = gwave / 24;       // 16-row block
  int cg    = gwave % 24;       // 64-col group
  v8f acc[4] = {};
  for (int kb = 0; kb < HID / 32; ++kb) {
    v16bf a = load_a_frag(insb, mblk * 16, kb, lane, HID);
#pragma unroll
    for (int n = 0; n < 4; ++n) {
      v16bf b = load_b_frag(wt, cg * 64 + n * 16, kb, lane, HID);
      acc[n] = wmma_bf16(a, b, acc[n]);
    }
  }
  int l16 = lane & 15, hi = (lane >> 4) & 1;
  size_t row = (size_t)mblk * 16 + hi * 8;
#pragma unroll
  for (int n = 0; n < 4; ++n) {
    int col = cg * 64 + n * 16 + l16;
    float bias = b_i[col];
    float* o = xp + row * (3 * HID) + col;
#pragma unroll
    for (int r = 0; r < 8; ++r) o[(size_t)r * (3 * HID)] = acc[n][r] + bias;
  }
}

// ---------------------------------------------------------------------------
// K5: persistent GRU scan. NWG blocks, each owns JW hidden units; recurrent
// weights live in LDS (192 rows x 512 k, bf16). Grid barrier per timestep.
__global__ void k_gru_scan(const float* __restrict__ xp,
                           const unsigned char* __restrict__ resets, // jax bool: 1 byte
                           const bf16* __restrict__ wrec,
                           const float* __restrict__ b_hn,
                           float* __restrict__ hstate,   // fp32 masked h [64,512]
                           bf16* __restrict__ hmask,     // bf16 mirror for WMMA A
                           unsigned int* __restrict__ cnt,
                           float* __restrict__ out) {
  extern __shared__ bf16 smem[];
  bf16*  wlds = smem;                                   // [192][WPITCH]
  float* S    = (float*)(smem + 3 * JW * WPITCH);       // [64][SPITCH]

  const int g    = blockIdx.x;       // hidden slice
  const int tid  = threadIdx.x;      // 256 threads = 8 waves
  const int lane = tid & 31;
  const int wave = tid >> 5;

  // --- stage this slice's recurrent weights into LDS (16B chunks) ---
  for (int idx = tid; idx < 3 * JW * (HID / 8); idx += 256) {
    int lr = idx >> 6;                 // local row 0..191
    int ch = idx & 63;                 // 16-byte chunk within row
    int gate = lr / JW;                // 0=r,1=z,2=n
    int grow = gate * HID + g * JW + (lr % JW);
    *(v8bf*)(wlds + (size_t)lr * WPITCH + ch * 8) =
        *((const v8bf*)(wrec + (size_t)grow * HID) + ch);
  }
  __syncthreads();

  const int m    = wave & 3;   // 16-row batch block
  const int half = wave >> 2;  // 6 col tiles each

  for (int t = 0; t < T_STEPS; ++t) {
    // ---- [64,512] x [512,192] slice GEMM ----
    v8f acc[6] = {};
    for (int kb = 0; kb < HID / 32; ++kb) {
      v16bf a = load_a_frag(hmask, m * 16, kb, lane, HID);
#pragma unroll
      for (int n = 0; n < 6; ++n) {
        v16bf b = load_b_frag(wlds, (half * 6 + n) * 16, kb, lane, WPITCH);
        acc[n] = wmma_bf16(a, b, acc[n]);
      }
    }
    // spill accumulators to LDS scratch (C layout: lane<16 -> M=r, else M=r+8)
    {
      int l16 = lane & 15, hi = (lane >> 4) & 1;
      int row = m * 16 + hi * 8;
#pragma unroll
      for (int n = 0; n < 6; ++n) {
        int col = (half * 6 + n) * 16 + l16;
#pragma unroll
        for (int r = 0; r < 8; ++r) S[(row + r) * SPITCH + col] = acc[n][r];
      }
    }
    __syncthreads();

    // ---- fused gate epilogue: 64 batch x 64 hidden ----
    {
      int jp = tid & 63;
      int b0 = tid >> 6;                 // 0..3
      int j  = g * JW + jp;
      float bh = b_hn[j];
#pragma unroll 4
      for (int bb = 0; bb < 16; ++bb) {
        int b = bb * 4 + b0;
        float hr = S[b * SPITCH + jp];
        float hz = S[b * SPITCH + 64 + jp];
        float hn = S[b * SPITCH + 128 + jp];
        const float* xprow = xp + ((size_t)t * BATCH + b) * (3 * HID);
        float xr = xprow[j], xz = xprow[HID + j], xn = xprow[2 * HID + j];
        float hprev = hstate[b * HID + j];           // already reset-masked
        float r = 1.f / (1.f + __expf(-(xr + hr)));
        float z = 1.f / (1.f + __expf(-(xz + hz)));
        float nn = tanhf(xn + r * (hn + bh));
        float hnew = (1.f - z) * nn + z * hprev;
        out[((size_t)t * BATCH + b) * HID + j] = hnew;
        float nxt = hnew;
        if (t + 1 < T_STEPS && resets[(t + 1) * BATCH + b]) nxt = 0.f;
        hstate[b * HID + j] = nxt;
        hmask[b * HID + j]  = (bf16)nxt;
      }
    }

    // ---- grid-wide barrier (monotonic count, agent-scope release/acquire) ----
    __threadfence();
    __syncthreads();
    if (tid == 0) {
      __hip_atomic_fetch_add(cnt, 1u, __ATOMIC_RELEASE, __HIP_MEMORY_SCOPE_AGENT);
      unsigned target = (unsigned)(t + 1) * NWG;
      while (__hip_atomic_load(cnt, __ATOMIC_ACQUIRE, __HIP_MEMORY_SCOPE_AGENT) < target) {
        __builtin_amdgcn_s_sleep(1);
      }
    }
    __syncthreads();
  }
}

// ---------------------------------------------------------------------------
extern "C" void kernel_launch(void* const* d_in, const int* in_sizes, int n_in,
                              void* d_out, int out_size, void* d_ws, size_t ws_size,
                              hipStream_t stream) {
  const float*         ins    = (const float*)d_in[0];
  const unsigned char* resets = (const unsigned char*)d_in[1];
  const float*         W_i    = (const float*)d_in[2];
  const float*         b_i    = (const float*)d_in[3];
  const float*         W_hrz  = (const float*)d_in[4];
  const float*         W_hn   = (const float*)d_in[5];
  const float*         b_hn   = (const float*)d_in[6];
  float*               out    = (float*)d_out;

  char* ws = (char*)d_ws;
  constexpr size_t XP_BYTES    = (size_t)T_STEPS * BATCH * 3 * HID * 4;  // 192 MB
  constexpr size_t INSB_BYTES  = (size_t)T_STEPS * BATCH * HID * 2;     // 32 MB
  constexpr size_t WTI_BYTES   = (size_t)3 * HID * HID * 2;             // 1.5 MB
  constexpr size_t WREC_BYTES  = (size_t)3 * HID * HID * 2;             // 1.5 MB
  constexpr size_t HST_BYTES   = (size_t)BATCH * HID * 4;
  constexpr size_t HMK_BYTES   = (size_t)BATCH * HID * 2;

  size_t off = 0;
  float* xp      = (float*)(ws + off);        off += XP_BYTES;
  bf16*  insb    = (bf16*)(ws + off);         off += INSB_BYTES;
  bf16*  wti     = (bf16*)(ws + off);         off += WTI_BYTES;
  bf16*  wrec    = (bf16*)(ws + off);         off += WREC_BYTES;
  float* hstate  = (float*)(ws + off);        off += HST_BYTES;
  bf16*  hmask   = (bf16*)(ws + off);         off += HMK_BYTES;
  unsigned int* cnt = (unsigned int*)(ws + off);

  constexpr int SMEM_BYTES = 3 * JW * WPITCH * 2 + BATCH * SPITCH * 4;  // 249856
  hipFuncSetAttribute(reinterpret_cast<const void*>(k_gru_scan),
                      hipFuncAttributeMaxDynamicSharedMemorySize, SMEM_BYTES);

  // K0: init state + barrier
  k_init<<<(BATCH * HID + 255) / 256, 256, 0, stream>>>(hstate, hmask, cnt);
  // K1: ins -> bf16
  {
    int n = T_STEPS * BATCH * HID;
    k_cvt_ins<<<(n + 255) / 256, 256, 0, stream>>>(ins, insb, n);
  }
  // K2/K3: weight images
  {
    int n = 3 * HID * HID;
    k_build_wti<<<(n + 255) / 256, 256, 0, stream>>>(W_i, wti);
    k_build_wrec<<<(n + 255) / 256, 256, 0, stream>>>(W_hrz, W_hn, wrec);
  }
  // K4: time-parallel input GEMM
  {
    int waves  = (T_STEPS * BATCH / 16) * (3 * HID / 64);  // 49152
    int blocks = waves / 8;                                 // 6144 (256 thr = 8 waves)
    k_gemm_xp<<<blocks, 256, 0, stream>>>(insb, wti, b_i, xp);
  }
  // K5: persistent sequential scan
  k_gru_scan<<<NWG, 256, SMEM_BYTES, stream>>>(xp, resets, wrec, b_hn,
                                               hstate, hmask, cnt, out);
}
